// Encoder_66039417143573
// MI455X (gfx1250) — matrix operand order repaired and checked by reading, not compiled, not run on hardware
//
#include <hip/hip_runtime.h>
#include <hip/hip_bf16.h>

#define BN_EPS 1e-5f

typedef __attribute__((ext_vector_type(4)))  __bf16 v4bf;
typedef __attribute__((ext_vector_type(8)))  __bf16 v8bf;
typedef __attribute__((ext_vector_type(16))) __bf16 v16bf;
typedef __attribute__((ext_vector_type(8)))  float  v8f;

__device__ __forceinline__ __bf16 f2bf(float f) { return (__bf16)f; }

// ---------------------------------------------------------------------------
// Tiled GEMM: C[M x Nout] = A[M x K] @ B[K x Nout] (+bias)(+accum)(+relu)
// fp32 in / fp32 out, bf16 WMMA (v_wmma_f32_16x16x32_bf16) compute.
// Block tile 128x128, K-step 32, 256 threads = 8 waves (2x4 wave grid),
// each wave computes 4x2 16x16 WMMA tiles. Double-buffered LDS with
// software-pipelined global fetch. Requires K%32==0, Nout%128==0.
// ---------------------------------------------------------------------------
#define TBM 128
#define TBN 128
#define TBK 32
#define SA  40   // LDS row stride (halves): padded vs 32 to dodge bank conflicts
#define SB  40
#define F_BIAS 1
#define F_RELU 2
#define F_ACC  4

__global__ __launch_bounds__(256)
void gemm_bf16_wmma(const float* __restrict__ A, const float* __restrict__ B,
                    const float* __restrict__ bias, float* __restrict__ C,
                    int M, int K, int Nout, int flags)
{
    __shared__ __align__(16) __bf16 ldsA[2][TBM * SA]; // [buf][row][k]
    __shared__ __align__(16) __bf16 ldsB[2][TBN * SB]; // [buf][n][k]  (B transposed)

    const int tid  = threadIdx.x;
    const int lane = tid & 31;
    const int wv   = tid >> 5;     // wave 0..7
    const int wm   = wv >> 2;      // 0..1  -> 64-row half
    const int wn   = wv & 3;       // 0..3  -> 32-col quarter
    const int half = lane >> 4;    // 0|1 (wave half)
    const int l16  = lane & 15;
    const int m0g  = blockIdx.x * TBM;
    const int n0g  = blockIdx.y * TBN;

    // ---- loop-invariant staging coordinates (per thread) ----
    int    rowA[4], colA[4], krB[4], colB[4];
    size_t gOffA[4], gOffB[4];
#pragma unroll
    for (int i = 0; i < 4; ++i) {
        const int f4 = tid + i * 256;          // 0..1023 float4 slots
        rowA[i] = f4 >> 3;                     // 0..127
        colA[i] = (f4 & 7) << 2;               // 0..28
        int gr  = m0g + rowA[i];
        if (gr > M - 1) gr = M - 1;            // clamp: padded rows never stored
        gOffA[i] = (size_t)gr * K + colA[i];
        krB[i]  = f4 >> 5;                     // 0..31
        colB[i] = (f4 & 31) << 2;              // 0..124
        gOffB[i] = (size_t)krB[i] * Nout + n0g + colB[i];
    }

    v8f zed = {};
    v8f acc[4][2];
#pragma unroll
    for (int mf = 0; mf < 4; ++mf)
#pragma unroll
        for (int nf = 0; nf < 2; ++nf) acc[mf][nf] = zed;

    // ---- prologue: fetch + stage tile 0 ----
    float4 ra[4], rb[4];
#pragma unroll
    for (int i = 0; i < 4; ++i) {
        ra[i] = *(const float4*)(A + gOffA[i]);
        rb[i] = *(const float4*)(B + gOffB[i]);
    }
#pragma unroll
    for (int i = 0; i < 4; ++i) {
        v4bf h;
        h[0] = f2bf(ra[i].x); h[1] = f2bf(ra[i].y);
        h[2] = f2bf(ra[i].z); h[3] = f2bf(ra[i].w);
        *(v4bf*)&ldsA[0][rowA[i] * SA + colA[i]] = h;
        ldsB[0][(colB[i] + 0) * SB + krB[i]] = f2bf(rb[i].x);
        ldsB[0][(colB[i] + 1) * SB + krB[i]] = f2bf(rb[i].y);
        ldsB[0][(colB[i] + 2) * SB + krB[i]] = f2bf(rb[i].z);
        ldsB[0][(colB[i] + 3) * SB + krB[i]] = f2bf(rb[i].w);
    }

    for (int kt = 0; kt < K; kt += TBK) {
        const int cur     = (kt >> 5) & 1;
        const bool hasNxt = (kt + TBK) < K;

        // issue next tile's global loads before the barrier (latency overlap)
        if (hasNxt) {
#pragma unroll
            for (int i = 0; i < 4; ++i) {
                ra[i] = *(const float4*)(A + gOffA[i] + (kt + TBK));
                rb[i] = *(const float4*)(B + gOffB[i] + (size_t)(kt + TBK) * Nout);
            }
        }
        __syncthreads();   // lds[cur] ready for all; lds[cur^1] drained by all

        // ---- fragments per ISA 16-bit layouts, 8 WMMAs ----
        v16bf af[4];
#pragma unroll
        for (int mf = 0; mf < 4; ++mf) {
            const int row = wm * 64 + mf * 16 + l16;
            const int c0  = half * 8;   // lanes 0-15: K 0-7/16-23 ; 16-31: 8-15/24-31
            const __bf16* ap = &ldsA[cur][row * SA];
            v8bf lo = *(const v8bf*)(ap + c0);
            v8bf hi = *(const v8bf*)(ap + c0 + 16);
            af[mf] = __builtin_shufflevector(lo, hi, 0,1,2,3,4,5,6,7,8,9,10,11,12,13,14,15);
        }
        v16bf bfm[2];
#pragma unroll
        for (int nf = 0; nf < 2; ++nf) {
            const int nrow = wn * 32 + nf * 16 + l16;
            const __bf16* bp = &ldsB[cur][nrow * SB + half * 8]; // 0-15: K 0-7/16-23 ; 16-31: K 8-15/24-31
            v8bf lo = *(const v8bf*)(bp);
            v8bf hi = *(const v8bf*)(bp + 16);
            bfm[nf] = __builtin_shufflevector(lo, hi, 0,1,2,3,4,5,6,7,8,9,10,11,12,13,14,15);
        }
#pragma unroll
        for (int mf = 0; mf < 4; ++mf)
#pragma unroll
            for (int nf = 0; nf < 2; ++nf)
                acc[mf][nf] = __builtin_amdgcn_wmma_f32_16x16x32_bf16(
                    false, af[mf], false, bfm[nf], (short)0, acc[mf][nf], false, false);

        // ---- stage next tile into the other buffer ----
        if (hasNxt) {
            const int nxt = cur ^ 1;
#pragma unroll
            for (int i = 0; i < 4; ++i) {
                v4bf h;
                h[0] = f2bf(ra[i].x); h[1] = f2bf(ra[i].y);
                h[2] = f2bf(ra[i].z); h[3] = f2bf(ra[i].w);
                *(v4bf*)&ldsA[nxt][rowA[i] * SA + colA[i]] = h;
                ldsB[nxt][(colB[i] + 0) * SB + krB[i]] = f2bf(rb[i].x);
                ldsB[nxt][(colB[i] + 1) * SB + krB[i]] = f2bf(rb[i].y);
                ldsB[nxt][(colB[i] + 2) * SB + krB[i]] = f2bf(rb[i].z);
                ldsB[nxt][(colB[i] + 3) * SB + krB[i]] = f2bf(rb[i].w);
            }
        }
    }

    // ---- epilogue: C/D layout -> row = base + half*8 + vgpr, col = l16 ----
#pragma unroll
    for (int mf = 0; mf < 4; ++mf) {
#pragma unroll
        for (int nf = 0; nf < 2; ++nf) {
            const int cidx  = n0g + wn * 32 + nf * 16 + l16;
            const int rbase = m0g + wm * 64 + mf * 16 + half * 8;
            const float bv  = (flags & F_BIAS) ? bias[cidx] : 0.f;
#pragma unroll
            for (int rr = 0; rr < 8; ++rr) {
                const int r = rbase + rr;
                if (r < M) {
                    float val = acc[mf][nf][rr] + bv;
                    const size_t o = (size_t)r * Nout + cidx;
                    if (flags & F_ACC)  val += C[o];
                    if (flags & F_RELU) val = fmaxf(val, 0.f);
                    C[o] = val;
                }
            }
        }
    }
}

// ---------------------------------------------------------------------------
// SAGE mean-aggregation: scatter-add h[src] rows into sums[dst] + degree count
// One thread per (edge, 4-float group): fully coalesced gather + f32 atomics.
// ---------------------------------------------------------------------------
__global__ __launch_bounds__(256)
void sage_scatter(const float* __restrict__ h, const long long* __restrict__ ei,
                  float* __restrict__ sums, float* __restrict__ cnts, int E)
{
    const int idx = blockIdx.x * 256 + threadIdx.x;
    if (idx >= E * 32) return;
    const int e = idx >> 5;
    const int f = (idx & 31) << 2;
    const long long s = ei[e];
    const long long d = ei[(size_t)E + e];
    const float4 v = *(const float4*)(h + (size_t)s * 128 + f);
    float* p = sums + (size_t)d * 128 + f;
    unsafeAtomicAdd(p + 0, v.x);
    unsafeAtomicAdd(p + 1, v.y);
    unsafeAtomicAdd(p + 2, v.z);
    unsafeAtomicAdd(p + 3, v.w);
    if (f == 0) unsafeAtomicAdd(cnts + d, 1.f);
}

__global__ __launch_bounds__(256)
void mean_div(float* __restrict__ aggr, const float* __restrict__ cnts, int n)
{
    const size_t idx = (size_t)blockIdx.x * 256 + threadIdx.x;
    if (idx >= (size_t)n * 128) return;
    const int row = (int)(idx >> 7);
    aggr[idx] = aggr[idx] / fmaxf(cnts[row], 1.f);
}

// ---------------------------------------------------------------------------
// BatchNorm batch statistics: per-column sum / sumsq, block-local accumulation
// + f32 atomics; columns coalesced across threads.
// ---------------------------------------------------------------------------
__global__ __launch_bounds__(256)
void bn_stats(const float* __restrict__ z, float* __restrict__ ssum,
              float* __restrict__ ssq, int n, int m)
{
    const int col = blockIdx.x * 256 + threadIdx.x;  // m % 256 == 0
    float s = 0.f, q = 0.f;
    for (int r = blockIdx.y; r < n; r += gridDim.y) {
        const float v = z[(size_t)r * m + col];
        s += v; q += v * v;
    }
    unsafeAtomicAdd(&ssum[col], s);
    unsafeAtomicAdd(&ssq[col], q);
}

__global__ __launch_bounds__(256)
void bn_apply_relu(float* __restrict__ z, const float* __restrict__ ssum,
                   const float* __restrict__ ssq, const float* __restrict__ g,
                   const float* __restrict__ be, int n, int m)
{
    const size_t idx = (size_t)blockIdx.x * 256 + threadIdx.x;
    if (idx >= (size_t)n * m) return;
    const int col = (int)(idx % m);
    const float inv_n = 1.f / (float)n;
    const float mean  = ssum[col] * inv_n;
    const float var   = ssq[col] * inv_n - mean * mean;
    const float v = (z[idx] - mean) * rsqrtf(var + BN_EPS) * g[col] + be[col];
    z[idx] = fmaxf(v, 0.f);
}

// ---------------------------------------------------------------------------
// Final logits: [N,256] @ [256,16] + bias. Tiny N-dim -> LDS-staged VALU dot.
// ---------------------------------------------------------------------------
__global__ __launch_bounds__(256)
void head_logits(const float* __restrict__ a, const float* __restrict__ w,
                 const float* __restrict__ b, float* __restrict__ out, int n)
{
    __shared__ float ws[256 * 16];
    __shared__ float bs[16];
    const int tid = threadIdx.x;
#pragma unroll
    for (int i = 0; i < 16; ++i) ws[tid + i * 256] = w[tid + i * 256];
    if (tid < 16) bs[tid] = b[tid];
    __syncthreads();
    const int row = blockIdx.x * 16 + (tid >> 4);
    const int c   = tid & 15;
    if (row < n) {
        float accv = bs[c];
        const float* ar = a + (size_t)row * 256;
#pragma unroll 8
        for (int k = 0; k < 256; ++k) accv += ar[k] * ws[k * 16 + c];
        out[(size_t)row * 16 + c] = accv;
    }
}

__global__ void zero_f32(float* p, size_t count)
{
    const size_t idx = (size_t)blockIdx.x * blockDim.x + threadIdx.x;
    if (idx < count) p[idx] = 0.f;
}

// ---------------------------------------------------------------------------
extern "C" void kernel_launch(void* const* d_in, const int* in_sizes, int n_in,
                              void* d_out, int out_size, void* d_ws, size_t ws_size,
                              hipStream_t stream)
{
    const float*     x   = (const float*)d_in[0];
    const long long* ei  = (const long long*)d_in[1];
    const float*     w1  = (const float*)d_in[2];
    const float*     b1  = (const float*)d_in[3];
    const float*     wl  = (const float*)d_in[4];
    const float*     bl  = (const float*)d_in[5];
    const float*     wr  = (const float*)d_in[6];
    const float*     hw1 = (const float*)d_in[7];
    const float*     hb1 = (const float*)d_in[8];
    const float*     g1  = (const float*)d_in[9];
    const float*     be1 = (const float*)d_in[10];
    const float*     hw2 = (const float*)d_in[11];
    const float*     hb2 = (const float*)d_in[12];
    const float*     g2  = (const float*)d_in[13];
    const float*     be2 = (const float*)d_in[14];
    const float*     hw3 = (const float*)d_in[15];
    const float*     hb3 = (const float*)d_in[16];
    const float*     g3  = (const float*)d_in[17];
    const float*     be3 = (const float*)d_in[18];
    const float*     hw4 = (const float*)d_in[19];
    const float*     hb4 = (const float*)d_in[20];

    const int N = in_sizes[0] / 64;
    const int E = in_sizes[1] / 2;

    float* out    = (float*)d_out;
    float* logits = out;                              // [N,16]
    float* pen    = out + (size_t)N * 16;             // [N,256]
    float* ofeat  = out + (size_t)N * (16 + 256);     // [N,128]

    float* ws   = (float*)d_ws;
    float* h0   = ws;                                 // [N,128]
    float* aggr = h0   + (size_t)N * 128;             // [N,128] (sums -> aggr in place)
    float* cnts = aggr + (size_t)N * 128;             // [N]
    float* z1   = cnts + (size_t)N;                   // [N,256]
    float* z2   = z1   + (size_t)N * 256;             // [N,1024]
    float* ssum = z2   + (size_t)N * 1024;            // [1024]
    float* ssq  = ssum + 1024;                        // [1024]

    const dim3 blk(256);
    const unsigned gm = (unsigned)((N + 127) / 128);
    auto b1d = [](size_t c) { return dim3((unsigned)((c + 255) / 256)); };

    // zero scatter targets (aggr + cnts contiguous) -- required every call
    zero_f32<<<b1d((size_t)N * 129), blk, 0, stream>>>(aggr, (size_t)N * 129);

    // conv1: h0 = relu(x @ w1 + b1)
    gemm_bf16_wmma<<<dim3(gm, 1), blk, 0, stream>>>(x, w1, b1, h0, N, 64, 128, F_BIAS | F_RELU);

    // SAGE mean aggregation
    sage_scatter<<<b1d((size_t)E * 32), blk, 0, stream>>>(h0, ei, aggr, cnts, E);
    mean_div<<<b1d((size_t)N * 128), blk, 0, stream>>>(aggr, cnts, N);

    // out_feat = aggr @ wl + bl + h0 @ wr
    gemm_bf16_wmma<<<dim3(gm, 1), blk, 0, stream>>>(aggr, wl, bl, ofeat, N, 128, 128, F_BIAS);
    gemm_bf16_wmma<<<dim3(gm, 1), blk, 0, stream>>>(h0, wr, nullptr, ofeat, N, 128, 128, F_ACC);

    // head L1: z1 = bn_relu(out_feat @ hw1 + hb1)
    gemm_bf16_wmma<<<dim3(gm, 2), blk, 0, stream>>>(ofeat, hw1, hb1, z1, N, 128, 256, F_BIAS);
    zero_f32<<<b1d(2048), blk, 0, stream>>>(ssum, 2048);
    bn_stats<<<dim3(1, 256), blk, 0, stream>>>(z1, ssum, ssq, N, 256);
    bn_apply_relu<<<b1d((size_t)N * 256), blk, 0, stream>>>(z1, ssum, ssq, g1, be1, N, 256);

    // head L2: z2 = bn_relu(z1 @ hw2 + hb2)
    gemm_bf16_wmma<<<dim3(gm, 8), blk, 0, stream>>>(z1, hw2, hb2, z2, N, 256, 1024, F_BIAS);
    zero_f32<<<b1d(2048), blk, 0, stream>>>(ssum, 2048);
    bn_stats<<<dim3(4, 256), blk, 0, stream>>>(z2, ssum, ssq, N, 1024);
    bn_apply_relu<<<b1d((size_t)N * 1024), blk, 0, stream>>>(z2, ssum, ssq, g2, be2, N, 1024);

    // head L3 -> penultimate (written straight into d_out, BN in place)
    gemm_bf16_wmma<<<dim3(gm, 2), blk, 0, stream>>>(z2, hw3, hb3, pen, N, 1024, 256, F_BIAS);
    zero_f32<<<b1d(2048), blk, 0, stream>>>(ssum, 2048);
    bn_stats<<<dim3(1, 256), blk, 0, stream>>>(pen, ssum, ssq, N, 256);
    bn_apply_relu<<<b1d((size_t)N * 256), blk, 0, stream>>>(pen, ssum, ssq, g3, be3, N, 256);

    // logits
    head_logits<<<dim3((unsigned)((N + 15) / 16)), blk, 0, stream>>>(pen, hw4, hb4, logits, N);

    (void)n_in; (void)out_size; (void)ws_size;
}
